// BsuiteRnnUnshared1Rnn_46952582480501
// MI455X (gfx1250) — matrix operand order repaired and checked by reading, not compiled, not run on hardware
//
#include <hip/hip_runtime.h>

// ---------------------------------------------------------------------------
// Problem constants (from reference)
// ---------------------------------------------------------------------------
#define TT 512
#define BB 256
#define OBS 128
#define HID 256
#define AA 16
#define RR 256
#define KEXT 288              // padded rnn input: 256 (body) + 16 (act) + 1 (rew) + 15 (zero)
#define MROWS (TT * 2 * BB)   // 262144 interleaved rows: row = t*512 + stream*256 + b
#define G3R 768
#define NBLK_SCAN 48

typedef __attribute__((ext_vector_type(16))) __bf16 v16bf;
typedef __attribute__((ext_vector_type(8)))  float  v8f;
typedef __attribute__((ext_vector_type(4)))  unsigned int u32x4;
typedef __attribute__((ext_vector_type(8)))  int i32x8;
typedef __attribute__((ext_vector_type(4)))  int i32x4;

union FragBF { v16bf v; unsigned int u[8]; u32x4 x4[2]; };

// ---------------------------------------------------------------------------
// small helpers
// ---------------------------------------------------------------------------
__device__ __forceinline__ unsigned short f2bf(float f) {
  union { float f; unsigned u; } x; x.f = f;
  unsigned u = x.u;
  u += 0x7FFFu + ((u >> 16) & 1u);        // round-to-nearest-even
  return (unsigned short)(u >> 16);
}
__device__ __forceinline__ float bf2f(unsigned short h) {
  union { unsigned u; float f; } x; x.u = ((unsigned)h) << 16;
  return x.f;
}
__device__ __forceinline__ float sigm(float x) { return 1.0f / (1.0f + __expf(-x)); }
__device__ __forceinline__ float tanh_fast(float x) {
  x = fminf(fmaxf(x, -10.0f), 10.0f);
  float e = __expf(2.0f * x);
  return (e - 1.0f) / (e + 1.0f);
}

// device-wide sense-reversal barrier (48 resident blocks only)
__device__ __forceinline__ void gridbar(unsigned int* cnt, volatile unsigned int* gen,
                                        unsigned int nb) {
  __threadfence();
  __syncthreads();
  if (threadIdx.x == 0) {
    unsigned int g = *gen;
    if (atomicAdd(cnt, 1u) == nb - 1u) {
      *cnt = 0u;
      __threadfence();
      *gen = g + 1u;
    } else {
      while (*gen == g) { __builtin_amdgcn_s_sleep(1); }
      __threadfence();
    }
  }
  __syncthreads();
}

// ---------------------------------------------------------------------------
// Tensor Data Mover staging: contiguous run of `nelem` bf16 elements
// global -> LDS, described as a 1-D tile (D# per CDNA5 ISA)
// This toolchain exposes the 6-arg builtin:
//   (v4u group0, v8i group1, v4i group2, v4i group3, v8i extra, i32 cpol)
// ---------------------------------------------------------------------------
#if __has_builtin(__builtin_amdgcn_tensor_load_to_lds) && \
    __has_builtin(__builtin_amdgcn_s_wait_tensorcnt)
#define USE_TDM 1
#endif

#ifdef USE_TDM
__device__ __forceinline__ void tdm_stage(const unsigned short* gsrc, void* lds_dst,
                                          int nelem /* 2-byte units, <= 65535 */) {
  unsigned long long ga = (unsigned long long)(uintptr_t)gsrc;
  unsigned ldsa = (unsigned)(uintptr_t)lds_dst;  // low 32 bits of generic = LDS offset
  u32x4 g0;
  g0[0] = 1u;                                           // count=1, user descriptor
  g0[1] = ldsa;                                         // lds_addr
  g0[2] = (unsigned)ga;                                 // global_addr[31:0]
  g0[3] = (unsigned)((ga >> 32) & 0x01FFFFFFu) | (2u << 30);  // addr[56:32] | type=2
  i32x8 g1;
  g1[0] = (int)(1u << 16);                              // data_size = 2 bytes
  g1[1] = (int)(((unsigned)nelem & 0xFFFFu) << 16);     // tensor_dim0[15:0]
  g1[2] = (int)((((unsigned)nelem >> 16) & 0xFFFFu)     // tensor_dim0[31:16]
                | (1u << 16));                          // tensor_dim1 = 1
  g1[3] = (int)(((unsigned)nelem & 0xFFFFu) << 16);     // tile_dim0 = nelem
  g1[4] = 1;                                            // tile_dim1 = 1
  g1[5] = (int)(unsigned)nelem;                         // tensor_dim0_stride lo
  g1[6] = 0;
  g1[7] = 0;
  i32x4 z4; z4[0] = 0; z4[1] = 0; z4[2] = 0; z4[3] = 0;
  i32x8 z8; z8[0] = 0; z8[1] = 0; z8[2] = 0; z8[3] = 0;
  z8[4] = 0; z8[5] = 0; z8[6] = 0; z8[7] = 0;
  __builtin_amdgcn_tensor_load_to_lds(g0, g1, z4, z4, z8, 0);
}
#endif

// stage `nelem` bf16 elements from W into LDS; whole block participates
__device__ __forceinline__ void stage_weights(unsigned int* sB, const unsigned short* W,
                                              int nelem, int tid) {
#ifdef USE_TDM
  if (tid < 32) {  // single wave issues the TDM op (EXEC-independent) and waits
    tdm_stage(W, sB, nelem);
    __builtin_amdgcn_s_wait_tensorcnt(0);
  }
#else
  const u32x4* g = (const u32x4*)W;
  u32x4* s = (u32x4*)sB;
  for (int i = tid; i < (nelem >> 3); i += 256) s[i] = g[i];
#endif
  __syncthreads();
}

// ---------------------------------------------------------------------------
// conversion / packing kernels
// ---------------------------------------------------------------------------
__global__ __launch_bounds__(256) void cvt_f32_bf16(const float* __restrict__ src,
                                                    unsigned short* __restrict__ dst, int n) {
  int i = blockIdx.x * 256 + threadIdx.x;
  if (i < n) dst[i] = f2bf(src[i]);
}

// W_ih [768,273] -> padded bf16 [768,288] (cols 273..287 zero)
__global__ __launch_bounds__(256) void wih_pad(const float* __restrict__ src,
                                               unsigned short* __restrict__ dst) {
  int i = blockIdx.x * 256 + threadIdx.x;
  if (i >= G3R * KEXT) return;
  int row = i / KEXT, col = i - row * KEXT;
  float v = (col < 273) ? src[row * 273 + col] : 0.0f;
  dst[i] = f2bf(v);
}

// observation [T*B,128] f32 -> interleaved duplicated bf16 [MROWS,128]
__global__ __launch_bounds__(256) void obs_dup(const float* __restrict__ obs,
                                               unsigned short* __restrict__ dst) {
  int e = blockIdx.x * 256 + threadIdx.x;   // < MROWS*128
  int g = e >> 7, k = e & 127;
  int t = g >> 9, b = g & 255;
  dst[e] = f2bf(obs[(t * BB + b) * OBS + k]);
}

// fill b_ext columns 256..287: action(16), reward(1), zero pad(15)
__global__ __launch_bounds__(256) void act_fill(const float* __restrict__ act,
                                                const float* __restrict__ rew,
                                                unsigned short* __restrict__ bext) {
  int e = blockIdx.x * 256 + threadIdx.x;   // < MROWS*32
  int g = e >> 5, j = e & 31;
  int t = g >> 9, b = g & 255;
  float v = 0.0f;
  if (j < 16)       v = act[(t * BB + b) * AA + j];
  else if (j == 16) v = rew[t * BB + b];
  bext[(size_t)g * KEXT + 256 + j] = f2bf(v);
}

// initialize h state [512,256] from h0_pi / h0_v
__global__ __launch_bounds__(256) void init_h(const float* __restrict__ h0p,
                                              const float* __restrict__ h0v,
                                              float* __restrict__ hf,
                                              unsigned short* __restrict__ hb) {
  int e = blockIdx.x * 256 + threadIdx.x;   // < 512*256
  int row = e >> 8, r = e & 255;
  float v = (row < 256) ? h0p[row * RR + r] : h0v[(row - 256) * RR + r];
  hf[e] = v;
  hb[e] = f2bf(v);
}

// ---------------------------------------------------------------------------
// generic bf16 WMMA GEMM:  C[M,N] = act( A[M,K] @ W[N,K]^T + bias )
// block = 256 thr (8 waves), tile 128x64; A frag double-buffered across k-steps
// ---------------------------------------------------------------------------
__global__ __launch_bounds__(256) void gemm_bf16(
    const unsigned short* __restrict__ A, int lda,
    const unsigned short* __restrict__ W0, const unsigned short* __restrict__ W1,
    const float* __restrict__ bias0, const float* __restrict__ bias1,
    unsigned short* __restrict__ C, int ldc,
    int K, int relu) {
  extern __shared__ __align__(16) unsigned char smem_raw[];
  unsigned int* sB = (unsigned int*)smem_raw;

  const int tid  = threadIdx.x;
  const int lane = tid & 31;
  const int wave = tid >> 5;
  const int n0   = blockIdx.x * 64;
  const int m0   = blockIdx.y * 128;
  const int strm = (blockIdx.y >> 1) & 1;   // 256-row chunks alternate pi/v
  const unsigned short* W = strm ? W1 : W0;
  const float* bias = strm ? bias1 : bias0;

  stage_weights(sB, W + (size_t)n0 * K, 64 * K, tid);

  const int rowA  = m0 + wave * 16 + (lane & 15);
  const int koffA = (lane >> 4) ? 8 : 0;
  const int koffB = (lane >> 4) ? 8 : 0;
  const int ncol  = lane & 15;
  const unsigned short* Arow = A + (size_t)rowA * lda;

  v8f acc[4];
#pragma unroll
  for (int q = 0; q < 4; ++q)
#pragma unroll
    for (int i = 0; i < 8; ++i) acc[q][i] = 0.0f;

  FragBF a;
  a.x4[0] = *(const u32x4*)(Arow + koffA);
  a.x4[1] = *(const u32x4*)(Arow + koffA + 16);

  for (int k0 = 0; k0 < K; k0 += 32) {
    const int kn = (k0 + 32 < K) ? (k0 + 32) : 0;   // prefetch next A frag
    FragBF an;
    an.x4[0] = *(const u32x4*)(Arow + kn + koffA);
    an.x4[1] = *(const u32x4*)(Arow + kn + koffA + 16);
#pragma unroll
    for (int q = 0; q < 4; ++q) {
      FragBF b;
      const unsigned int* bp = sB + (((q * 16 + ncol) * K + k0 + koffB) >> 1);
      b.x4[0] = *(const u32x4*)(bp);
      b.x4[1] = *(const u32x4*)(bp + 8);
      acc[q] = __builtin_amdgcn_wmma_f32_16x16x32_bf16(
          false, a.v, false, b.v, (short)0, acc[q], false, false);
    }
    a = an;
  }

  const int rbase = m0 + wave * 16 + ((lane >> 4) << 3);
#pragma unroll
  for (int q = 0; q < 4; ++q) {
    const int n = n0 + q * 16 + ncol;
    const float bv = bias[n];
#pragma unroll
    for (int v = 0; v < 8; ++v) {
      float x = acc[q][v] + bv;
      if (relu) x = fmaxf(x, 0.0f);
      C[(size_t)(rbase + v) * ldc + n] = f2bf(x);
    }
  }
}

// ---------------------------------------------------------------------------
// persistent GRU scan: 48 blocks, W_hh slice resident in LDS across all T steps
// ---------------------------------------------------------------------------
__global__ __launch_bounds__(256) void gru_scan(
    const unsigned short* __restrict__ Whh,   // [768,256] bf16
    const float* __restrict__ b_hh,           // [768]
    const unsigned short* __restrict__ gx,    // [T*512,768] bf16
    float* __restrict__ gh,                   // [512,768] f32 scratch
    float* __restrict__ hf,                   // [512,256] f32 state
    unsigned short* __restrict__ hb,          // [512,256] bf16 state
    unsigned short* __restrict__ y,           // [T*512,256] bf16 relu(h)
    float* __restrict__ hT_pi, float* __restrict__ hT_v,
    unsigned int* __restrict__ bar_cnt, volatile unsigned int* bar_gen) {
  extern __shared__ __align__(16) unsigned char smem_raw[];
  unsigned int* sB = (unsigned int*)smem_raw;

  const int tid  = threadIdx.x;
  const int lane = tid & 31;
  const int wave = tid >> 5;
  const int blk  = blockIdx.x;
  const int ct   = blk % 12;     // 12 col tiles * 64 = 768
  const int rt   = blk / 12;     // 4 row tiles * 128 = 512
  const int n0   = ct * 64;
  const int m0   = rt * 128;

  stage_weights(sB, Whh + (size_t)n0 * RR, 64 * RR, tid);

  const int rowA  = m0 + wave * 16 + (lane & 15);
  const int koffA = (lane >> 4) ? 8 : 0;
  const int koffB = (lane >> 4) ? 8 : 0;
  const int ncol  = lane & 15;
  const int rbase = m0 + wave * 16 + ((lane >> 4) << 3);
  const unsigned short* Arow = hb + (size_t)rowA * RR;

  for (int t = 0; t < TT; ++t) {
    // ---- phase A: gh = hb @ Whh^T + b_hh
    v8f acc[4];
#pragma unroll
    for (int q = 0; q < 4; ++q)
#pragma unroll
      for (int i = 0; i < 8; ++i) acc[q][i] = 0.0f;

    FragBF a;
    a.x4[0] = *(const u32x4*)(Arow + koffA);
    a.x4[1] = *(const u32x4*)(Arow + koffA + 16);
    for (int k0 = 0; k0 < RR; k0 += 32) {
      const int kn = (k0 + 32 < RR) ? (k0 + 32) : 0;
      FragBF an;
      an.x4[0] = *(const u32x4*)(Arow + kn + koffA);
      an.x4[1] = *(const u32x4*)(Arow + kn + koffA + 16);
#pragma unroll
      for (int q = 0; q < 4; ++q) {
        FragBF b;
        const unsigned int* bp = sB + (((q * 16 + ncol) * RR + k0 + koffB) >> 1);
        b.x4[0] = *(const u32x4*)(bp);
        b.x4[1] = *(const u32x4*)(bp + 8);
        acc[q] = __builtin_amdgcn_wmma_f32_16x16x32_bf16(
            false, a.v, false, b.v, (short)0, acc[q], false, false);
      }
      a = an;
    }
#pragma unroll
    for (int q = 0; q < 4; ++q) {
      const int n = n0 + q * 16 + ncol;
      const float bv = b_hh[n];
#pragma unroll
      for (int v = 0; v < 8; ++v)
        gh[(size_t)(rbase + v) * G3R + n] = acc[q][v] + bv;
    }
    gridbar(bar_cnt, bar_gen, NBLK_SCAN);

    // ---- phase B: gates + state update
    const size_t gbase = (size_t)t * 512;
    for (int e = blk * 256 + tid; e < 512 * 256; e += NBLK_SCAN * 256) {
      const int row = e >> 8;
      const int r   = e & 255;
      const size_t grow = gbase + row;
      const float xr = bf2f(gx[grow * G3R + r]);
      const float xz = bf2f(gx[grow * G3R + 256 + r]);
      const float xn = bf2f(gx[grow * G3R + 512 + r]);
      const float hr = gh[(size_t)row * G3R + r];
      const float hz = gh[(size_t)row * G3R + 256 + r];
      const float hn = gh[(size_t)row * G3R + 512 + r];
      const float rg = sigm(xr + hr);
      const float zg = sigm(xz + hz);
      const float ng = tanh_fast(xn + rg * hn);
      const float hprev = hf[e];
      const float hnew  = (1.0f - zg) * ng + zg * hprev;
      hf[e] = hnew;
      hb[e] = f2bf(hnew);
      y[grow * RR + r] = f2bf(fmaxf(hnew, 0.0f));
    }
    gridbar(bar_cnt, bar_gen, NBLK_SCAN);
  }

  // final hidden states
  for (int e = blk * 256 + tid; e < 512 * 256; e += NBLK_SCAN * 256) {
    const int row = e >> 8;
    const int r   = e & 255;
    const float hv = hf[e];
    if (row < 256) hT_pi[row * RR + r] = hv;
    else           hT_v[(row - 256) * RR + r] = hv;
  }
}

// ---------------------------------------------------------------------------
// heads
// ---------------------------------------------------------------------------
__global__ __launch_bounds__(256) void pi_head(const unsigned short* __restrict__ y,
                                               const float* __restrict__ Wp,
                                               const float* __restrict__ bp,
                                               float* __restrict__ out) {
  int L = blockIdx.x * 256 + threadIdx.x;   // < T*B*16
  int rowp = L >> 4, a = L & 15;
  int t = rowp >> 8, b = rowp & 255;
  const unsigned short* yr = y + ((size_t)t * 512 + b) * RR;  // pi rows: first 256 of chunk
  const float* w = Wp + a * RR;
  float s = bp[a];
  for (int k = 0; k < RR; k += 2) {
    unsigned u = *(const unsigned*)(yr + k);
    s += bf2f((unsigned short)u) * w[k] + bf2f((unsigned short)(u >> 16)) * w[k + 1];
  }
  float m = s;
#pragma unroll
  for (int msk = 8; msk >= 1; msk >>= 1) m = fmaxf(m, __shfl_xor(m, msk, 32));
  float e = __expf(s - m);
  float tot = e;
#pragma unroll
  for (int msk = 8; msk >= 1; msk >>= 1) tot += __shfl_xor(tot, msk, 32);
  out[(size_t)rowp * AA + a] = e / tot;
}

__global__ __launch_bounds__(256) void v_head(const unsigned short* __restrict__ y,
                                              const float* __restrict__ Wv,
                                              const float* __restrict__ bv,
                                              float* __restrict__ out) {
  int row = blockIdx.x * 256 + threadIdx.x;   // < T*B
  int t = row >> 8, b = row & 255;
  const unsigned short* yr = y + ((size_t)t * 512 + 256 + b) * RR;  // v rows: second 256
  float s = bv[0];
  for (int k = 0; k < RR; k += 2) {
    unsigned u = *(const unsigned*)(yr + k);
    s += bf2f((unsigned short)u) * Wv[k] + bf2f((unsigned short)(u >> 16)) * Wv[k + 1];
  }
  out[row] = s;
}

// ---------------------------------------------------------------------------
// host launcher
// ---------------------------------------------------------------------------
extern "C" void kernel_launch(void* const* d_in, const int* in_sizes, int n_in,
                              void* d_out, int out_size, void* d_ws, size_t ws_size,
                              hipStream_t stream) {
  const float* observation = (const float*)d_in[0];
  const float* prev_action = (const float*)d_in[1];
  const float* prev_reward = (const float*)d_in[2];
  const float* h0_pi = (const float*)d_in[3];
  const float* h0_v  = (const float*)d_in[4];
  const float* W1_pi = (const float*)d_in[5];
  const float* b1_pi = (const float*)d_in[6];
  const float* W2_pi = (const float*)d_in[7];
  const float* b2_pi = (const float*)d_in[8];
  const float* W1_v  = (const float*)d_in[9];
  const float* b1_v  = (const float*)d_in[10];
  const float* W2_v  = (const float*)d_in[11];
  const float* b2_v  = (const float*)d_in[12];
  const float* W_ih  = (const float*)d_in[13];
  const float* b_ih  = (const float*)d_in[14];
  const float* W_hh  = (const float*)d_in[15];
  const float* b_hh  = (const float*)d_in[16];
  const float* W_pi_head = (const float*)d_in[17];
  const float* b_pi_head = (const float*)d_in[18];
  const float* W_v_head  = (const float*)d_in[19];
  const float* b_v_head  = (const float*)d_in[20];

  // output layout: pi[T*B*16] | v[T*B] | hT_pi[B*R] | hT_v[B*R]
  float* out_pi   = (float*)d_out;
  float* out_v    = out_pi + (size_t)TT * BB * AA;
  float* out_hTpi = out_v + (size_t)TT * BB;
  float* out_hTv  = out_hTpi + (size_t)BB * RR;

  // workspace carve-out
  char* ws = (char*)d_ws;
  size_t off = 0;
  auto take = [&](size_t bytes) -> char* {
    char* p = ws + off;
    off = (off + bytes + 255) & ~(size_t)255;
    return p;
  };
  unsigned int*   bar    = (unsigned int*)take(256);
  unsigned short* w1pi_b = (unsigned short*)take((size_t)HID * OBS * 2);
  unsigned short* w1v_b  = (unsigned short*)take((size_t)HID * OBS * 2);
  unsigned short* w2pi_b = (unsigned short*)take((size_t)HID * HID * 2);
  unsigned short* w2v_b  = (unsigned short*)take((size_t)HID * HID * 2);
  unsigned short* wih_b  = (unsigned short*)take((size_t)G3R * KEXT * 2);
  unsigned short* whh_b  = (unsigned short*)take((size_t)G3R * RR * 2);
  float*          hf     = (float*)take((size_t)512 * RR * 4);
  unsigned short* hb     = (unsigned short*)take((size_t)512 * RR * 2);
  float*          gh     = (float*)take((size_t)512 * G3R * 4);
  unsigned short* obs_b  = (unsigned short*)take((size_t)MROWS * OBS * 2);
  unsigned short* h1_b   = (unsigned short*)take((size_t)MROWS * HID * 2);
  unsigned short* bext_b = (unsigned short*)take((size_t)MROWS * KEXT * 2);
  unsigned short* gx_b   = (unsigned short*)take((size_t)MROWS * G3R * 2);
  unsigned short* y_b    = obs_b;  // alias: obs/h1 dead before scan writes y

  (void)hipMemsetAsync(bar, 0, 256, stream);

  // weight conversions
  cvt_f32_bf16<<<(HID * OBS + 255) / 256, 256, 0, stream>>>(W1_pi, w1pi_b, HID * OBS);
  cvt_f32_bf16<<<(HID * OBS + 255) / 256, 256, 0, stream>>>(W1_v, w1v_b, HID * OBS);
  cvt_f32_bf16<<<(HID * HID + 255) / 256, 256, 0, stream>>>(W2_pi, w2pi_b, HID * HID);
  cvt_f32_bf16<<<(HID * HID + 255) / 256, 256, 0, stream>>>(W2_v, w2v_b, HID * HID);
  cvt_f32_bf16<<<(G3R * RR + 255) / 256, 256, 0, stream>>>(W_hh, whh_b, G3R * RR);
  wih_pad<<<(G3R * KEXT + 255) / 256, 256, 0, stream>>>(W_ih, wih_b);

  // activation packing
  obs_dup<<<((size_t)MROWS * OBS) / 256, 256, 0, stream>>>(observation, obs_b);
  act_fill<<<((size_t)MROWS * 32) / 256, 256, 0, stream>>>(prev_action, prev_reward, bext_b);
  init_h<<<(512 * RR) / 256, 256, 0, stream>>>(h0_pi, h0_v, hf, hb);

  // MLP layer 1: [MROWS,128] x [256,128]^T -> h1 (relu)
  gemm_bf16<<<dim3(HID / 64, MROWS / 128), 256, 64 * OBS * 2, stream>>>(
      obs_b, OBS, w1pi_b, w1v_b, b1_pi, b1_v, h1_b, HID, OBS, 1);
  // MLP layer 2: [MROWS,256] x [256,256]^T -> bext cols 0..255 (relu)
  gemm_bf16<<<dim3(HID / 64, MROWS / 128), 256, 64 * HID * 2, stream>>>(
      h1_b, HID, w2pi_b, w2v_b, b2_pi, b2_v, bext_b, KEXT, HID, 1);
  // GRU input projection: [MROWS,288] x [768,288]^T -> gx (+ b_ih)
  gemm_bf16<<<dim3(G3R / 64, MROWS / 128), 256, 64 * KEXT * 2, stream>>>(
      bext_b, KEXT, wih_b, wih_b, b_ih, b_ih, gx_b, G3R, KEXT, 0);

  // sequential GRU scan (persistent, W_hh resident in LDS, device-wide barrier)
  gru_scan<<<NBLK_SCAN, 256, 64 * RR * 2, stream>>>(
      whh_b, b_hh, gx_b, gh, hf, hb, y_b, out_hTpi, out_hTv, bar, bar + 64);

  // heads
  pi_head<<<((size_t)TT * BB * AA) / 256, 256, 0, stream>>>(y_b, W_pi_head, b_pi_head, out_pi);
  v_head<<<((size_t)TT * BB) / 256, 256, 0, stream>>>(y_b, W_v_head, b_v_head, out_v);

  (void)in_sizes; (void)n_in; (void)out_size; (void)ws_size;
}
